// TestPointLSTM_69148973465804
// MI455X (gfx1250) — matrix operand order repaired and checked by reading, not compiled, not run on hardware
//
#include <hip/hip_runtime.h>
#include <cstddef>

typedef __attribute__((ext_vector_type(16))) __bf16 v16bf;
typedef __attribute__((ext_vector_type(8)))  float  v8f;

// Global-address-space pointer types: keep A-fragment loads as global_load_b128
// (a laundered generic pointer would lower to flat_load_b128, which also burns
// DScnt and the LDS path per the CDNA5 ISA).
typedef const __bf16 __attribute__((address_space(1)))* gbf16p;
typedef const v16bf  __attribute__((address_space(1)))* gv16bfp;

constexpr int B_   = 4;
constexpr int C_   = 132;
constexpr int T_   = 32;
constexpr int N_   = 64;
constexpr int HID_ = 256;
constexpr int OFF_ = 4;
constexpr int K_   = 16;
constexpr int CIN_ = C_ + OFF_ + HID_;   // 392
constexpr int KC_  = 13;                 // K-dim chunks of 32 (padded to 416)
constexpr int NC_  = 2;                  // points per chunk -> 32 tile-tasks = 1/wave
constexpr int MT_  = HID_ / 16;          // 16 hidden tiles

// CDNA5 16-bit A-matrix 16x32 fragment layout: (lane, elem) -> (M, K)
__device__ __forceinline__ void a_map(int lane, int e, int& m, int& k) {
  int hi = (lane >> 4) & 1;
  m = lane & 15;
  int v = e >> 1, p = e & 1;
  if (v < 4) k = hi * 8 + v * 2 + p;            // VGPR0-3: K 0..7 / 8..15
  else       k = 16 + hi * 8 + (v - 4) * 2 + p; // VGPR4-7: K 16..23 / 24..31
}

__global__ void pack_W(const float* __restrict__ W, __bf16* __restrict__ Wp, int total) {
  int i = blockIdx.x * blockDim.x + threadIdx.x;
  if (i >= total) return;
  int e    = i & 15;
  int lane = (i >> 4) & 31;
  int kc   = (i >> 9) % KC_;
  int rt   = i / (KC_ * 512);
  int m, k;
  a_map(lane, e, m, k);
  int row = rt * 16 + m;
  int col = kc * 32 + k;
  float v = (col < CIN_) ? W[row * CIN_ + col] : 0.0f;
  Wp[i] = (__bf16)v;
}

__device__ __forceinline__ float sigmoidf(float x) { return 1.0f / (1.0f + __expf(-x)); }

__global__ __launch_bounds__(1024) void plstm(
    const float* __restrict__ inputs, const float* __restrict__ bias,
    const __bf16* __restrict__ Wp, float* __restrict__ out, int* __restrict__ inds)
{
  __shared__ __align__(128) unsigned char smem[163840];       // exactly 160 KB
  __bf16 (*c_buf)[HID_][N_] = (__bf16 (*)[HID_][N_])(smem);            // 65536 B
  __bf16 (*h_buf)[HID_][N_] = (__bf16 (*)[HID_][N_])(smem + 65536);    // 65536 B
  unsigned char* pool = smem + 131072;                                 // 26624 B
  __bf16* Bfrag = (__bf16*)pool;            // NC_*KC_*512 bf16 = 26624 B
  float*  dmat  = (float*)pool;             // 64*64 f32 (aliased, disjoint phases)
  int   (*idx_lds)[K_] = (int (*)[K_])(smem + 157696);                 // 4096 B
  float (*pt)[N_]      = (float (*)[N_])(smem + 161792);               // 1024 B
  float (*ppos)[N_]    = (float (*)[N_])(smem + 162816);               // 1024 B

  const int b    = blockIdx.x;
  const int tid  = threadIdx.x;
  const int lane = tid & 31;
  const int wid  = tid >> 5;

  // Per-wave constants: each wave owns hidden tile `mtile` and point-slot `pc0`.
  const int mtile = wid & 15;
  const int pc0   = wid >> 4;
  const int col   = lane & 15;                 // neighbor slot in D tile
  const int mh    = (lane >> 4) * 8;           // C/D layout: lanes16-31 hold M=8..15

  gbf16p aB0 = (gbf16p)(Wp + (size_t)(0 * MT_ + mtile) * (KC_ * 512) + lane * 16);
  gbf16p aB1 = (gbf16p)(Wp + (size_t)(1 * MT_ + mtile) * (KC_ * 512) + lane * 16);
  gbf16p aB2 = (gbf16p)(Wp + (size_t)(2 * MT_ + mtile) * (KC_ * 512) + lane * 16);
  gbf16p aB3 = (gbf16p)(Wp + (size_t)(3 * MT_ + mtile) * (KC_ * 512) + lane * 16);
  const __bf16* bB = Bfrag + pc0 * (KC_ * 512) + lane * 16;

  float bR[4][8];
  #pragma unroll
  for (int q = 0; q < 4; ++q)
    #pragma unroll
    for (int j = 0; j < 8; ++j)
      bR[q][j] = bias[q * HID_ + mtile * 16 + mh + j];

  for (int i = tid; i < 2 * HID_ * N_; i += 1024) {
    ((__bf16*)c_buf)[i] = (__bf16)0.0f;
    ((__bf16*)h_buf)[i] = (__bf16)0.0f;
  }
  for (int i = tid; i < OFF_ * N_; i += 1024) {
    int ch = i / N_, n = i % N_;
    ppos[ch][n] = inputs[(((size_t)b * C_ + ch) * T_ + 0) * N_ + n];
  }
  __syncthreads();

  int cur = 0;
  for (int t = 0; t < T_; ++t) {
    // stage current positions pt = inputs[b, :4, t, :]
    for (int i = tid; i < OFF_ * N_; i += 1024) {
      int ch = i / N_, n = i % N_;
      pt[ch][n] = inputs[(((size_t)b * C_ + ch) * T_ + t) * N_ + n];
    }
    __syncthreads();
    // 64x64 squared-distance matrix over first 3 pos channels
    for (int i = tid; i < N_ * N_; i += 1024) {
      int n = i >> 6, m = i & 63;
      float d = 0.f;
      #pragma unroll
      for (int ch = 0; ch < 3; ++ch) {
        float df = pt[ch][n] - ppos[ch][m];
        d += df * df;
      }
      dmat[i] = d;
    }
    __syncthreads();
    // top-K smallest (ties -> lower index), one wave per point, 2 points/wave
    for (int rep = 0; rep < 2; ++rep) {
      int n = wid + rep * 32;
      float v0 = dmat[n * 64 + lane];
      float v1 = dmat[n * 64 + lane + 32];
      bool u0 = false, u1 = false;
      for (int k = 0; k < K_; ++k) {
        float a0 = u0 ? 3.4e38f : v0;
        float a1 = u1 ? 3.4e38f : v1;
        float bv; int bi;
        if (a0 <= a1) { bv = a0; bi = lane; } else { bv = a1; bi = lane + 32; }
        #pragma unroll
        for (int off = 16; off >= 1; off >>= 1) {
          float ov = __shfl_xor(bv, off);
          int   oi = __shfl_xor(bi, off);
          if (ov < bv || (ov == bv && oi < bi)) { bv = ov; bi = oi; }
        }
        if (lane == 0) {
          idx_lds[n][k] = bi;
          inds[(((size_t)b * T_ + t) * N_ + n) * K_ + k] = bi;
        }
        if (bi == lane) u0 = true;
        else if (bi == lane + 32) u1 = true;
      }
    }
    __syncthreads();   // also fences dmat -> Bfrag aliasing

    for (int chunk = 0; chunk < N_ / NC_; ++chunk) {
      // build bf16 B fragments (comb = [x ; pos_g - pt ; h_gather]) for NC_ points
      for (int i = tid; i < NC_ * KC_ * 512; i += 1024) {
        int e   = i & 15;
        int ln  = (i >> 4) & 31;
        int kc  = (i >> 9) % KC_;
        int pc  = i / (KC_ * 512);
        int n   = chunk * NC_ + pc;
        int bcol = ln & 15;                        // neighbor slot 0..15
        int r    = kc * 32 + ((e < 8) ? ((ln >> 4) * 8 + e) : (16 + (ln >> 4) * 8 + (e - 8))); // K row (cin index)
        float v;
        if (r < C_) {
          v = inputs[(((size_t)b * C_ + r) * T_ + t) * N_ + n];
        } else if (r < C_ + OFF_) {
          int ch = r - C_;
          int nb = idx_lds[n][bcol];
          v = ppos[ch][nb] - pt[ch][n];
        } else if (r < CIN_) {
          int hr = r - (C_ + OFF_);
          int nb = idx_lds[n][bcol];
          v = (float)h_buf[cur][hr][nb];
        } else {
          v = 0.0f;                               // K padding 392..415
        }
        Bfrag[i] = (__bf16)v;
      }
      __syncthreads();

      // one (16 hidden rows x one point's 16 neighbors) tile-task per wave.
      // Launder the A base pointers (keeps loads loop-variant, so LICM cannot
      // hoist 52 fragments into registers and spill) while preserving
      // addrspace(1) so they stay global_load_b128.
      gbf16p pA0 = aB0;
      gbf16p pA1 = aB1;
      gbf16p pA2 = aB2;
      gbf16p pA3 = aB3;
      asm volatile("" : "+v"(pA0), "+v"(pA1), "+v"(pA2), "+v"(pA3));

      int n = chunk * NC_ + pc0;
      v8f acc0 = {}, acc1 = {}, acc2 = {}, acc3 = {};
      #pragma unroll 2
      for (int kc = 0; kc < KC_; ++kc) {
        v16bf bf = *(const v16bf*)(bB  + kc * 512);
        v16bf a0 = *(gv16bfp)(pA0 + kc * 512);
        v16bf a1 = *(gv16bfp)(pA1 + kc * 512);
        v16bf a2 = *(gv16bfp)(pA2 + kc * 512);
        v16bf a3 = *(gv16bfp)(pA3 + kc * 512);
        acc0 = __builtin_amdgcn_wmma_f32_16x16x32_bf16(false, a0, false, bf, (short)0, acc0, false, false);
        acc1 = __builtin_amdgcn_wmma_f32_16x16x32_bf16(false, a1, false, bf, (short)0, acc1, false, false);
        acc2 = __builtin_amdgcn_wmma_f32_16x16x32_bf16(false, a2, false, bf, (short)0, acc2, false, false);
        acc3 = __builtin_amdgcn_wmma_f32_16x16x32_bf16(false, a3, false, bf, (short)0, acc3, false, false);
      }
      // elementwise LSTM + max over the 16 neighbor columns (shuffle butterfly)
      int nb = idx_lds[n][col];
      #pragma unroll
      for (int j = 0; j < 8; ++j) {
        int hrow = mtile * 16 + mh + j;
        float gi = acc0[j] + bR[0][j];
        float gf = acc1[j] + bR[1][j];
        float go = acc2[j] + bR[2][j];
        float gg = acc3[j] + bR[3][j];
        float cg = (float)c_buf[cur][hrow][nb];
        float ck = sigmoidf(gf) * cg + sigmoidf(gi) * tanhf(gg);
        float hk = sigmoidf(go) * tanhf(ck);
        float cm = ck, hm = hk;
        #pragma unroll
        for (int off = 8; off >= 1; off >>= 1) {
          hm = fmaxf(hm, __shfl_xor(hm, off));
          cm = fmaxf(cm, __shfl_xor(cm, off));
        }
        if (col == 0) {
          h_buf[cur ^ 1][hrow][n] = (__bf16)hm;
          c_buf[cur ^ 1][hrow][n] = (__bf16)cm;
          out[(((size_t)b * HID_ + hrow) * T_ + t) * N_ + n] = hm;
        }
      }
      __syncthreads();
    }
    // carry: prev_pos = pt
    for (int i = tid; i < OFF_ * N_; i += 1024) {
      int ch = i / N_, n = i % N_;
      ppos[ch][n] = pt[ch][n];
    }
    cur ^= 1;
    __syncthreads();
  }
}

extern "C" void kernel_launch(void* const* d_in, const int* in_sizes, int n_in,
                              void* d_out, int out_size, void* d_ws, size_t ws_size,
                              hipStream_t stream) {
  const float* inputs = (const float*)d_in[0];
  // d_in[1] = offsets (unused by the reference forward)
  const float* W    = (const float*)d_in[2];
  const float* bias = (const float*)d_in[3];

  __bf16* Wp  = (__bf16*)d_ws;                              // 1024 x 416 bf16 = 832 KB
  float*  out = (float*)d_out;                               // (B,HID,T,N) fp32
  int*    inds = (int*)(out + (size_t)B_ * HID_ * T_ * N_);  // (B,T,N,K) int32 bits

  int total = (4 * HID_ / 16) * KC_ * 512;                   // 425984 packed elems
  pack_W<<<(total + 255) / 256, 256, 0, stream>>>(W, Wp, total);
  plstm<<<B_, 1024, 0, stream>>>(inputs, bias, Wp, out, inds);
}
